// GeometricStructureEmbedding_81054622810268
// MI455X (gfx1250) — compile-verified, matrix operand order checked by reading, not run on hardware
//
#include <hip/hip_runtime.h>
#include <hip/hip_bf16.h>
#include <float.h>

typedef __attribute__((ext_vector_type(16))) _Float16 v16h;
typedef __attribute__((ext_vector_type(8)))  _Float16 v8h;
typedef __attribute__((ext_vector_type(8)))  float    v8f;
typedef __attribute__((ext_vector_type(4)))  float    v4f;
typedef __attribute__((ext_vector_type(2)))  float    v2f;

#define NPTS 8192
#define KNN  35
#define HID  128
#define ROWS (NPTS * KNN)                  // 286720 (n,k) rows
#define EMB_ELEMS ((size_t)ROWS * HID)     // fp32 output elems before idx tail
#define FACTOR_A   3.8197186342054885f     // 180/(15*pi) = 12/pi
#define INV_SIGMA_D 5.0f                   // 1/0.2
#define LOG1E4_64  0.1439115683f           // ln(10000)/64

// ---------------------------------------------------------------------------
// Kernel 0: convert weights fp32 -> f16 (workspace)
// ---------------------------------------------------------------------------
__global__ void convert_w(const float* __restrict__ Wd, const float* __restrict__ Wa,
                          _Float16* __restrict__ Wdh, _Float16* __restrict__ Wah) {
    int i = blockIdx.x * blockDim.x + threadIdx.x;
    if (i < HID * HID) {
        Wdh[i] = (_Float16)Wd[i];
        Wah[i] = (_Float16)Wa[i];
    }
}

// ---------------------------------------------------------------------------
// Kernel 1: WMMA-based brute-force KNN.
// Gram matrix via V_WMMA_F32_16X16X4_F32 (K=4, xyz + dead lane).
// Ranking key per candidate c for query q:  |c|^2 - 2*(c.q)   (|q|^2 const).
// One wave = 16 queries (B fragment, loop-invariant). Candidates staged in
// LDS as float4 + a contiguous |c|^2 array; 16 candidates per WMMA.
// C layout: lane = query column, VGPR v = candidate row v + 8*(lane>=16).
// Per-lane top-35 lists live in LDS (stride 35 floats: gcd(35,64)=1 ->
// conflict-free banks); halves merged in-place from LDS by lanes 0-15.
// ---------------------------------------------------------------------------
#define KNN_STAGE_BYTES (NPTS * 16 + NPTS * 4)               // 160 KB
#define KNN_LIST_FLOATS (256 * KNN)                          // 8960 per array
#define KNN_LDS_BYTES   (KNN_STAGE_BYTES + 2 * KNN_LIST_FLOATS * 4)  // 230 KB

__global__ void __launch_bounds__(256, 1)
knn_wmma_kernel(const float* __restrict__ pts,
                float* __restrict__ dist_out,
                int* __restrict__ idx_out) {
    extern __shared__ char smem[];
    float* sp4 = (float*)smem;                    // NPTS * float4
    float* scn = (float*)(smem + NPTS * 16);      // NPTS * float
    float* lbd = (float*)(smem + KNN_STAGE_BYTES);                        // 256*35
    int*   lbi = (int*)(smem + KNN_STAGE_BYTES + KNN_LIST_FLOATS * 4);    // 256*35

    // ---- stage all candidate points (+norms) into LDS ----
    for (int i = threadIdx.x; i < NPTS; i += blockDim.x) {
        float x = pts[i * 3 + 0];
        float y = pts[i * 3 + 1];
        float z = pts[i * 3 + 2];
        float cn = x * x + y * y + z * z;
        v4f p = {x, y, z, cn};
        *(v4f*)(sp4 + i * 4) = p;
        scn[i] = cn;
    }

    // ---- init my LDS top-k list ----
    const int tslot = threadIdx.x * KNN;
    for (int j = 0; j < KNN; ++j) { lbd[tslot + j] = FLT_MAX; lbi[tslot + j] = 0; }
    __syncthreads();

    const int lane = threadIdx.x & 31;
    const int wave = threadIdx.x >> 5;
    const int m    = lane & 15;
    const int kc   = lane >> 4;                   // candidate-row half
    const int qbase = (blockIdx.x * 8 + wave) * 16;
    const int q     = qbase + m;

    // query coords (each lane loads its own column's query)
    const float qx = pts[q * 3 + 0];
    const float qy = pts[q * 3 + 1];
    const float qz = pts[q * 3 + 2];
    const float qnorm = qx * qx + qy * qy + qz * qz;

    // B fragment (4x16 f32): lanes 0-15 hold (K0,K1)=(x,y), lanes 16-31 (K2,K3)=(z,0)
    v2f B;
    B[0] = kc ? qz : qx;
    B[1] = kc ? 0.0f : qy;

    float worst = FLT_MAX;
    int   wslot = 0;

    for (int cb = 0; cb < NPTS; cb += 16) {
        // A fragment (16x4 f32): lanes 0-15 -> (x,y) of cand m; lanes 16-31 -> (z, junk)
        // junk K3 multiplies B's K3 == 0, so it never contributes.
        v2f A = *(const v2f*)(sp4 + (cb + m) * 4 + kc * 2);
        v8f C = {};
        C = __builtin_amdgcn_wmma_f32_16x16x4_f32(
            false, A, false, B, (short)0, C, false, false);

        // candidate norms for my 8 rows (contiguous -> two b128 loads)
        v4f cn0 = *(const v4f*)(scn + cb + kc * 8);
        v4f cn1 = *(const v4f*)(scn + cb + kc * 8 + 4);
#pragma unroll
        for (int v = 0; v < 8; ++v) {
            float cnv = (v < 4) ? cn0[v] : cn1[v - 4];
            float val = fmaf(-2.0f, C[v], cnv);   // |c|^2 - 2*dot
            if (val < worst) {
                lbd[tslot + wslot] = val;
                lbi[tslot + wslot] = cb + kc * 8 + v;
                worst = -FLT_MAX;
                for (int s = 0; s < KNN; ++s) {
                    float f = lbd[tslot + s];
                    if (f > worst) { worst = f; wslot = s; }
                }
            }
        }
    }

    // ---- merge the two half-lists per query straight out of LDS ----
    __syncthreads();
    if (lane < 16) {
        const int sA = (wave * 32 + m) * KNN;        // my half (kc=0 lane)
        const int sB = (wave * 32 + m + 16) * KNN;   // partner half (kc=1 lane)
        for (int s = 0; s < KNN; ++s) {
            float bv = FLT_MAX;
            int   bid = 0x7fffffff;
            int   bj = sA;
            for (int j = 0; j < 2 * KNN; ++j) {
                int addr = (j < KNN) ? (sA + j) : (sB + (j - KNN));
                float v = lbd[addr];
                int   id = lbi[addr];
                if (v < bv || (v == bv && id < bid)) { bv = v; bid = id; bj = addr; }
            }
            dist_out[q * KNN + s] = fmaxf(qnorm + bv, 0.0f);
            idx_out[q * KNN + s]  = bid;
            lbd[bj] = FLT_MAX;                        // consume slot
        }
    }
}

// ---------------------------------------------------------------------------
// Kernel 2: geometry + sinusoidal embedding + WMMA GEMMs + anchor max + bias.
// 8 waves/block; wave handles 16 rows x 128 cols of output.
// A fragments built directly in the CDNA5 16-bit A-matrix lane layout and
// kept register-resident; __launch_bounds__(256,1) grants the full VGPR file.
// B fragments (f16 weights) streamed from LDS (ds_load_b128 x2 each).
// ---------------------------------------------------------------------------
__global__ void __launch_bounds__(256, 1)
embed_kernel(const float* __restrict__ pts,
             const float* __restrict__ bdv,
             const float* __restrict__ bav,
             const _Float16* __restrict__ Wdh,
             const _Float16* __restrict__ Wah,
             const float* __restrict__ dists,
             const int* __restrict__ knn,
             float* __restrict__ out) {
    extern __shared__ _Float16 lds[];            // 2 * 128*128 halfs = 64 KB
    _Float16* sWd = lds;
    _Float16* sWa = lds + HID * HID;
    for (int i = threadIdx.x; i < HID * HID; i += blockDim.x) {
        sWd[i] = Wdh[i];
        sWa[i] = Wah[i];
    }
    __syncthreads();

    const int lane = threadIdx.x & 31;
    const int wave = threadIdx.x >> 5;
    const int tile = blockIdx.x * 8 + wave;      // one 16-row M-tile per wave
    const int rbase = tile * 16;
    const int m  = lane & 15;                    // my row within the tile
    const int kc = lane >> 4;                    // K-half (A/B lane layout)

    const int r = rbase + m;
    const int n = r / KNN;
    const int k = r - n * KNN;

    // ---- geometry: 1 distance index + 3 angular indices for row r ----
    float x[4];
    x[0] = sqrtf(fmaxf(dists[r], 1e-8f)) * INV_SIGMA_D;

    const float pnx = pts[n * 3 + 0], pny = pts[n * 3 + 1], pnz = pts[n * 3 + 2];
    const int   jk  = knn[n * KNN + k];
    const float pkx = pts[jk * 3 + 0], pky = pts[jk * 3 + 1], pkz = pts[jk * 3 + 2];
    const float ax = pkx - pnx, ay = pky - pny, az = pkz - pnz;   // anchor diff
#pragma unroll
    for (int a = 0; a < 3; ++a) {
        int ja = knn[n * KNN + a];
        float rx = pts[ja * 3 + 0] - pkx;
        float ry = pts[ja * 3 + 1] - pky;
        float rz = pts[ja * 3 + 2] - pkz;
        float cx = ry * az - rz * ay;
        float cy = rz * ax - rx * az;
        float cz = rx * ay - ry * ax;
        float sv = sqrtf(cx * cx + cy * cy + cz * cz);
        float cv = rx * ax + ry * ay + rz * az;
        x[1 + a] = atan2f(sv, cv) * FACTOR_A;
    }

    // ---- build A fragments: A[path][kstep], path 0=dist, 1..3=anchors ----
    // lane half h <-> K = (h<8 ? h : h+8) + 8*kc ; element = emb[32*s + K]
    // emb[2i]=sin(x*div_i), emb[2i+1]=cos(x*div_i), div_i = 1e4^(-i/64)
    v16h A[4][4];
#pragma unroll
    for (int s = 0; s < 4; ++s) {
        const int bi0 = 16 * s + kc * 4;
#pragma unroll
        for (int p = 0; p < 8; ++p) {
            const int i = bi0 + ((p < 4) ? p : (4 + p));   // 16s+kc*4+{0..3, 8..11}
            const float dv = __expf(-LOG1E4_64 * (float)i);
#pragma unroll
            for (int t = 0; t < 4; ++t) {
                float sv, cv;
                __sincosf(x[t] * dv, &sv, &cv);
                A[t][s][2 * p]     = (_Float16)sv;
                A[t][s][2 * p + 1] = (_Float16)cv;
            }
        }
    }

    // ---- GEMM over 8 column tiles of 16; K = 128 in 4 steps of 32 ----
    const int laddr_base = m * HID + kc * 8;     // per-lane half-offset into W rows
    float* outbase = out + (size_t)rbase * HID;

    for (int nt = 0; nt < 8; ++nt) {
        v8f accd = {};
        v8f a0 = {}, a1 = {}, a2 = {};
        const int ncol = nt * 16 + m;            // B/C lane column
        const int nbase = nt * 16 * HID + laddr_base;
#pragma unroll
        for (int s = 0; s < 4; ++s) {
            const int addr = nbase + 32 * s;
            // B fragment for W_d
            v8h lo = *(const v8h*)(sWd + addr);
            v8h hi = *(const v8h*)(sWd + addr + 16);
            v16h Bd = __builtin_shufflevector(lo, hi,
                0, 1, 2, 3, 4, 5, 6, 7, 8, 9, 10, 11, 12, 13, 14, 15);
            accd = __builtin_amdgcn_wmma_f32_16x16x32_f16(
                false, A[0][s], false, Bd, (short)0, accd, false, false);
            // B fragment for W_a, reused by 3 anchor WMMAs
            v8h lo2 = *(const v8h*)(sWa + addr);
            v8h hi2 = *(const v8h*)(sWa + addr + 16);
            v16h Ba = __builtin_shufflevector(lo2, hi2,
                0, 1, 2, 3, 4, 5, 6, 7, 8, 9, 10, 11, 12, 13, 14, 15);
            a0 = __builtin_amdgcn_wmma_f32_16x16x32_f16(
                false, A[1][s], false, Ba, (short)0, a0, false, false);
            a1 = __builtin_amdgcn_wmma_f32_16x16x32_f16(
                false, A[2][s], false, Ba, (short)0, a1, false, false);
            a2 = __builtin_amdgcn_wmma_f32_16x16x32_f16(
                false, A[3][s], false, Ba, (short)0, a2, false, false);
        }
        const float biasd = bdv[ncol];
        const float biasa = bav[ncol];
#pragma unroll
        for (int v = 0; v < 8; ++v) {
            const int row = v + kc * 8;          // C-matrix lane layout
            float mv = fmaxf(fmaxf(a0[v], a1[v]), a2[v]);
            // streaming (non-temporal) store: write-once output, keep L2 for
            // the hot weight/KNN working set
            __builtin_nontemporal_store(accd[v] + biasd + mv + biasa,
                                        outbase + (size_t)row * HID + ncol);
        }
    }
}

// ---------------------------------------------------------------------------
extern "C" void kernel_launch(void* const* d_in, const int* in_sizes, int n_in,
                              void* d_out, int out_size, void* d_ws, size_t ws_size,
                              hipStream_t stream) {
    (void)in_sizes; (void)n_in; (void)out_size; (void)ws_size;
    const float* pts = (const float*)d_in[0];
    const float* Wd  = (const float*)d_in[1];
    const float* bd  = (const float*)d_in[2];
    const float* Wa  = (const float*)d_in[3];
    const float* ba  = (const float*)d_in[4];

    float* out = (float*)d_out;
    // output tail: knn_idx (int bits) right after the fp32 embedding block
    int* idx_out = (int*)d_out + EMB_ELEMS;

    // workspace layout: [dist f32 x ROWS][W_d f16 x 16384][W_a f16 x 16384]
    float*    ws_dist = (float*)d_ws;
    _Float16* Wdh = (_Float16*)((char*)d_ws + (size_t)ROWS * sizeof(float));
    _Float16* Wah = Wdh + HID * HID;

    convert_w<<<(HID * HID + 255) / 256, 256, 0, stream>>>(Wd, Wa, Wdh, Wah);

    // 64 blocks x 8 waves x 16 queries = 8192 queries
    knn_wmma_kernel<<<NPTS / 128, 256, KNN_LDS_BYTES, stream>>>(
        pts, ws_dist, idx_out);

    const size_t lds_bytes = 2 * (size_t)HID * HID * sizeof(_Float16); // 64 KB
    embed_kernel<<<ROWS / 16 / 8, 256, lds_bytes, stream>>>(
        pts, bd, ba, Wdh, Wah, ws_dist, idx_out, out);
}
